// BaseLayer_48369921688085
// MI455X (gfx1250) — matrix operand order repaired and checked by reading, compiled, not run on hardware
//
#include <hip/hip_runtime.h>
#include <math.h>

// ---------------- problem constants ----------------
#define E_ 8
#define D_ 1024
#define F_ 4096
#define T_ 8192            // 4 * 2048 tokens
#define LN_EPS 1e-5f
#define TILES_M (T_ / 32)  // 256 M-tiles of 32 tokens

typedef __attribute__((ext_vector_type(16))) __bf16 v16bf;
typedef __attribute__((ext_vector_type(8)))  __bf16 v8bf;
typedef __attribute__((ext_vector_type(8)))  float  v8f;

// ---------------- workspace layout (bytes) ----------------
#define WS_COUNTS 0
#define WS_ORDER  1024
#define WS_ALPHA  (WS_ORDER + (size_t)E_ * T_ * 4)
#define WS_XN     ((size_t)1 << 20)
#define WS_H      (WS_XN + (size_t)2 * T_ * D_)

// =====================================================================
// Kernel 0: zero the per-expert token counters (ws is not re-poisoned)
// =====================================================================
__global__ void zero_counts_kernel(int* counts) {
    if (threadIdx.x < E_) counts[threadIdx.x] = 0;
}

// =====================================================================
// Kernel 1: routing (argmax over x@centroids^T), sigmoid gate, LayerNorm
// + assigned expert's gamma/beta, emit bf16 xn and expert token lists.
// =====================================================================
__global__ __launch_bounds__(128)
void routing_ln_kernel(const float* __restrict__ x,
                       const float* __restrict__ cent,   // [E,D]
                       const float* __restrict__ ln_g,   // [E,D]
                       const float* __restrict__ ln_b,   // [E,D]
                       int* __restrict__ counts,
                       int* __restrict__ order,          // [E,T]
                       float* __restrict__ alpha,        // [T]
                       __bf16* __restrict__ xn)          // [T,D]
{
    const int t   = blockIdx.x;
    const int tid = threadIdx.x;
    const int i0  = tid * 8;

    __shared__ float red[10][128];
    __shared__ int   sbest;
    __shared__ float smu, srinv;

    float xv[8];
    const float4* xp = (const float4*)(x + (size_t)t * D_ + i0);
    float4 a0 = xp[0], a1 = xp[1];
    xv[0] = a0.x; xv[1] = a0.y; xv[2] = a0.z; xv[3] = a0.w;
    xv[4] = a1.x; xv[5] = a1.y; xv[6] = a1.z; xv[7] = a1.w;

    float s = 0.f, ss = 0.f;
    float dot[E_];
#pragma unroll
    for (int e = 0; e < E_; ++e) dot[e] = 0.f;
#pragma unroll
    for (int j = 0; j < 8; ++j) { s += xv[j]; ss += xv[j] * xv[j]; }
#pragma unroll
    for (int e = 0; e < E_; ++e) {
        const float* cp = cent + (size_t)e * D_ + i0;
#pragma unroll
        for (int j = 0; j < 8; ++j) dot[e] += xv[j] * cp[j];
    }

    red[0][tid] = s;
    red[1][tid] = ss;
#pragma unroll
    for (int e = 0; e < E_; ++e) red[2 + e][tid] = dot[e];

    for (int st = 64; st >= 1; st >>= 1) {
        __syncthreads();
        if (tid < st)
            for (int a = 0; a < 10; ++a) red[a][tid] += red[a][tid + st];
    }
    __syncthreads();

    if (tid == 0) {
        float mu  = red[0][0] * (1.0f / D_);
        float var = red[1][0] * (1.0f / D_) - mu * mu;
        int   best = 0;
        float bv   = red[2][0];
        for (int e = 1; e < E_; ++e) {
            float v = red[2 + e][0];
            if (v > bv) { bv = v; best = e; }   // first-max, matches jnp.argmax
        }
        smu   = mu;
        srinv = rsqrtf(var + LN_EPS);
        sbest = best;
        alpha[t] = 1.0f / (1.0f + expf(-bv));
        int slot = atomicAdd(&counts[best], 1);
        order[best * T_ + slot] = t;
    }
    __syncthreads();

    const int   e  = sbest;
    const float mu = smu, ri = srinv;
    const float* gp = ln_g + (size_t)e * D_ + i0;
    const float* bp = ln_b + (size_t)e * D_ + i0;
    v8bf o;
#pragma unroll
    for (int j = 0; j < 8; ++j)
        o[j] = (__bf16)(((xv[j] - mu) * ri) * gp[j] + bp[j]);
    *(v8bf*)(xn + (size_t)t * D_ + i0) = o;
}

// =====================================================================
// Grouped-GEMM block: 32 tokens x 64 outputs, 128 threads = 4 waves.
// K is consumed in 64-wide slabs (2 fragment K-slices per buffer):
//   per iteration each wave issues 4 x v_wmma_f32_16x16x32_bf16
//   (2 K-slices x 2 N-tiles) with a single block barrier.
// Double-buffered LDS (24KB). Fragment pre-swizzle (ISA 05_wmma 7.12.2)
// so every lane reads each v16bf fragment as two contiguous ds_load_b128.
// Cooperative loader: lr = tid>>2 (tile row), lq = tid&3 (8-elem quarter):
//   A slot: [(lr&15) + ((lq&1)<<4)][(lq>>1)<<3]
//   B slot: [(lr&15) + ((lq&1)<<4)][(lq>>1)<<3]
// =====================================================================

#define GEMM_PRELUDE(KDIM, A_PTR, A_STRIDE, B_PTR, B_STRIDE)                  \
    const int tid = threadIdx.x;                                              \
    if (tid < 32) {                                                           \
        int r = m0 + tid;                                                     \
        toks[tid] = order[e * T_ + ((r < count) ? r : m0)];                   \
    }                                                                         \
    __syncthreads();                                                          \
    const int lane = tid & 31;                                                \
    const int wave = tid >> 5;                                                \
    const int wm   = wave & 1;                                                \
    const int wn0  = (wave >> 1) << 1;                                        \
    const int lr = tid >> 2;                                                  \
    const int lq = tid & 3;                                                   \
    const int tokr = toks[lr];                                                \
    const int aRow = (lr & 15) + ((lq & 1) << 4);                             \
    const int aCol = (lq >> 1) << 3;                                          \
    const int bRow = (lr & 15) + ((lq & 1) << 4);                             \
    const int bCol = (lq >> 1) << 3;                                          \
    const int rT   = lr >> 4;                                                 \
    const __bf16* aSrc  = (A_PTR) + (size_t)tokr * (A_STRIDE) + (lq << 3);    \
    const float*  bSrc0 = (B_PTR) + (size_t)(n0 + lr) * (B_STRIDE) + (lq << 3); \
    const float*  bSrc1 = bSrc0 + (size_t)32 * (B_STRIDE);                    \
    v8f acc0 = {}, acc1 = {};                                                 \
    v8bf areg[2], b0reg[2], b1reg[2];                                         \
    auto cvt8 = [](float4 p, float4 q) {                                      \
        v8bf r;                                                               \
        r[0] = (__bf16)p.x; r[1] = (__bf16)p.y;                               \
        r[2] = (__bf16)p.z; r[3] = (__bf16)p.w;                               \
        r[4] = (__bf16)q.x; r[5] = (__bf16)q.y;                               \
        r[6] = (__bf16)q.z; r[7] = (__bf16)q.w;                               \
        return r;                                                             \
    };                                                                        \
    auto loadSlab = [&](int k0) {                                             \
        _Pragma("unroll")                                                     \
        for (int ks = 0; ks < 2; ++ks) {                                      \
            const int kk = k0 + 32 * ks;                                      \
            areg[ks] = *(const v8bf*)(aSrc + kk);                             \
            float4 p0 = *(const float4*)(bSrc0 + kk);                         \
            float4 p1 = *(const float4*)(bSrc0 + kk + 4);                     \
            float4 q0 = *(const float4*)(bSrc1 + kk);                         \
            float4 q1 = *(const float4*)(bSrc1 + kk + 4);                     \
            b0reg[ks] = cvt8(p0, p1);                                         \
            b1reg[ks] = cvt8(q0, q1);                                         \
        }                                                                     \
    };                                                                        \
    auto storeSlab = [&](int buf) {                                           \
        _Pragma("unroll")                                                     \
        for (int ks = 0; ks < 2; ++ks) {                                      \
            *(v8bf*)&ldsA[buf][ks][rT][aRow][aCol]     = areg[ks];            \
            *(v8bf*)&ldsB[buf][ks][rT][bRow][bCol]     = b0reg[ks];           \
            *(v8bf*)&ldsB[buf][ks][2 + rT][bRow][bCol] = b1reg[ks];           \
        }                                                                     \
    };                                                                        \
    loadSlab(0);                                                              \
    storeSlab(0);                                                             \
    __syncthreads();                                                          \
    int buf = 0;                                                              \
    for (int k0 = 0; k0 < (KDIM); k0 += 64) {                                 \
        const bool last = (k0 + 64 >= (KDIM));                                \
        if (!last) {                                                          \
            loadSlab(k0 + 64);                                                \
            __builtin_prefetch(bSrc0 + k0 + 128, 0, 0);                       \
        }                                                                     \
        _Pragma("unroll")                                                     \
        for (int ks = 0; ks < 2; ++ks) {                                      \
            v16bf afrag = *(const v16bf*)&ldsA[buf][ks][wm][lane][0];         \
            v16bf bf0   = *(const v16bf*)&ldsB[buf][ks][wn0][lane][0];        \
            v16bf bf1   = *(const v16bf*)&ldsB[buf][ks][wn0 + 1][lane][0];    \
            acc0 = __builtin_amdgcn_wmma_f32_16x16x32_bf16(                   \
                false, afrag, false, bf0, (short)0, acc0, false, false);      \
            acc1 = __builtin_amdgcn_wmma_f32_16x16x32_bf16(                   \
                false, afrag, false, bf1, (short)0, acc1, false, false);      \
        }                                                                     \
        if (!last) storeSlab(buf ^ 1);                                        \
        __syncthreads();                                                      \
        buf ^= 1;                                                             \
    }

// Kernel 2: h = relu(xn @ w1[e]^T + b1[e]) for each expert's token list.
__global__ __launch_bounds__(128)
void ffn1_kernel(const __bf16* __restrict__ xn,   // [T,D] bf16
                 const float*  __restrict__ w1,   // [E,F,D]
                 const float*  __restrict__ b1,   // [E,F]
                 const int*    __restrict__ counts,
                 const int*    __restrict__ order,
                 __bf16*       __restrict__ h)    // [T,F] bf16
{
    const int e     = blockIdx.x / TILES_M;
    const int mt    = blockIdx.x % TILES_M;
    const int n0    = blockIdx.y * 64;
    const int count = counts[e];
    const int m0    = mt * 32;
    if (m0 >= count) return;                      // uniform per block

    __shared__ __attribute__((aligned(32))) __bf16 ldsA[2][2][2][32][16];
    __shared__ __attribute__((aligned(32))) __bf16 ldsB[2][2][4][32][16];
    __shared__ int toks[32];

    GEMM_PRELUDE(D_, xn, D_, w1 + (size_t)e * F_ * D_, D_)

    // C/D layout: VGPR r, lane l -> M = r + 8*(l>>4), N = l&15
    const int nglob0 = n0 + (wn0 << 4) + (lane & 15);
    const int nglob1 = nglob0 + 16;
    const int mhalf  = (lane >> 4) << 3;
    const float bias0 = b1[(size_t)e * F_ + nglob0];
    const float bias1 = b1[(size_t)e * F_ + nglob1];
#pragma unroll
    for (int r = 0; r < 8; ++r) {
        int row  = (wm << 4) + mhalf + r;
        int grow = m0 + row;
        if (grow < count) {
            int tok = toks[row];
            float v0 = acc0[r] + bias0;
            float v1 = acc1[r] + bias1;
            v0 = v0 > 0.f ? v0 : 0.f;
            v1 = v1 > 0.f ? v1 : 0.f;
            h[(size_t)tok * F_ + nglob0] = (__bf16)v0;
            h[(size_t)tok * F_ + nglob1] = (__bf16)v1;
        }
    }
}

// Kernel 3: out = x + alpha * (h @ w2[e]^T + b2[e])
__global__ __launch_bounds__(128)
void ffn2_kernel(const __bf16* __restrict__ hin,  // [T,F] bf16
                 const float*  __restrict__ w2,   // [E,D,F]
                 const float*  __restrict__ b2,   // [E,D]
                 const float*  __restrict__ x,    // [T,D]
                 const float*  __restrict__ alpha,// [T]
                 const int*    __restrict__ counts,
                 const int*    __restrict__ order,
                 float*        __restrict__ out)  // [T,D]
{
    const int e     = blockIdx.x / TILES_M;
    const int mt    = blockIdx.x % TILES_M;
    const int n0    = blockIdx.y * 64;
    const int count = counts[e];
    const int m0    = mt * 32;
    if (m0 >= count) return;

    __shared__ __attribute__((aligned(32))) __bf16 ldsA[2][2][2][32][16];
    __shared__ __attribute__((aligned(32))) __bf16 ldsB[2][2][4][32][16];
    __shared__ int toks[32];

    GEMM_PRELUDE(F_, hin, F_, w2 + (size_t)e * D_ * F_, F_)

    const int dglob0 = n0 + (wn0 << 4) + (lane & 15);
    const int dglob1 = dglob0 + 16;
    const int mhalf  = (lane >> 4) << 3;
    const float bias0 = b2[(size_t)e * D_ + dglob0];
    const float bias1 = b2[(size_t)e * D_ + dglob1];
#pragma unroll
    for (int r = 0; r < 8; ++r) {
        int row  = (wm << 4) + mhalf + r;
        int grow = m0 + row;
        if (grow < count) {
            int tok = toks[row];
            float a  = alpha[tok];
            float v0 = acc0[r] + bias0;
            float v1 = acc1[r] + bias1;
            out[(size_t)tok * D_ + dglob0] = x[(size_t)tok * D_ + dglob0] + a * v0;
            out[(size_t)tok * D_ + dglob1] = x[(size_t)tok * D_ + dglob1] + a * v1;
        }
    }
}

// =====================================================================
extern "C" void kernel_launch(void* const* d_in, const int* in_sizes, int n_in,
                              void* d_out, int out_size, void* d_ws, size_t ws_size,
                              hipStream_t stream) {
    const float* x    = (const float*)d_in[0];
    const float* cent = (const float*)d_in[1];
    const float* ln_g = (const float*)d_in[2];
    const float* ln_b = (const float*)d_in[3];
    const float* w1   = (const float*)d_in[4];
    const float* b1   = (const float*)d_in[5];
    const float* w2   = (const float*)d_in[6];
    const float* b2   = (const float*)d_in[7];
    float* out = (float*)d_out;

    char*   ws     = (char*)d_ws;
    int*    counts = (int*)(ws + WS_COUNTS);
    int*    order  = (int*)(ws + WS_ORDER);
    float*  alpha  = (float*)(ws + WS_ALPHA);
    __bf16* xn     = (__bf16*)(ws + WS_XN);
    __bf16* h      = (__bf16*)(ws + WS_H);

    hipLaunchKernelGGL(zero_counts_kernel, dim3(1), dim3(32), 0, stream, counts);
    hipLaunchKernelGGL(routing_ln_kernel, dim3(T_), dim3(128), 0, stream,
                       x, cent, ln_g, ln_b, counts, order, alpha, xn);
    hipLaunchKernelGGL(ffn1_kernel, dim3(E_ * TILES_M, F_ / 64), dim3(128), 0, stream,
                       xn, w1, b1, counts, order, h);
    hipLaunchKernelGGL(ffn2_kernel, dim3(E_ * TILES_M, D_ / 64), dim3(128), 0, stream,
                       h, w2, b2, x, alpha, counts, order, out);
}
